// ShiftedWindowAttention_5111011082413
// MI455X (gfx1250) — compile-verified, matrix operand order checked by reading, not run on hardware
//
#include <hip/hip_runtime.h>
#include <hip/hip_bf16.h>

// ---------------------------------------------------------------------------
// Shifted-window attention (Swin): B=8, L=4096, C=512, NH=16, D=32, WS=64,
// SHIFT=32 -> 512 windows of 64 tokens. bf16 WMMA with f32 accumulation.
// ---------------------------------------------------------------------------

typedef __bf16 bf16x8  __attribute__((ext_vector_type(8)));
typedef __bf16 bf16x16 __attribute__((ext_vector_type(16)));
typedef float  f32x8   __attribute__((ext_vector_type(8)));
typedef float  f32x4   __attribute__((ext_vector_type(4)));

union Frag {
    struct { bf16x8 lo, hi; } h;
    bf16x16 v;
};

__device__ __forceinline__ f32x8 wmma_bf16(const Frag& a, const Frag& b, f32x8 c) {
    return __builtin_amdgcn_wmma_f32_16x16x32_bf16(
        /*neg_a=*/false, a.v, /*neg_b=*/false, b.v,
        /*c_mod=*/(short)0, c, /*reuse_a=*/false, /*reuse_b=*/false);
}

#define QSCALE 0.17677669529663687f  /* 32^-0.5 */

// dword counts of the packed-weight regions in d_ws
#define WQ_DWORDS (16 * 96 * 32 * 8)   // qkv: 16 ksteps x 96 col-tiles x 32 lanes x 8 dwords
#define WP_DWORDS (16 * 32 * 32 * 8)   // proj: 16 ksteps x 32 col-tiles

// ---------------------------------------------------------------------------
// Pack weights into B-fragment order (bf16), folding q-scale into Wq.
// B fragment of W^T for tile (kt, tj): lane holds col n = tj*16 + lane%16,
// dword v holds K pair k = kt*32 + (lane>=16 ? 8:0) + (v>=4 ? 16:0) + (v&3)*2.
// ---------------------------------------------------------------------------
__global__ __launch_bounds__(256) void pack_weights_kernel(
        const float* __restrict__ qkv_w,   // [1536,512] row-major
        const float* __restrict__ proj_w,  // [512,512]
        uint32_t* __restrict__ wq,
        uint32_t* __restrict__ wp) {
    int idx = blockIdx.x * 256 + threadIdx.x;
    if (idx < WQ_DWORDS) {
        int v = idx & 7, lane = (idx >> 3) & 31, t = idx >> 8;
        int tj = t % 96, kt = t / 96;
        int n = tj * 16 + (lane & 15);
        int k = kt * 32 + ((lane & 16) ? 8 : 0) + ((v & 4) ? 16 : 0) + ((v & 3) << 1);
        float s = (n < 512) ? QSCALE : 1.0f;   // fold q scaling into Wq
        union { __bf16 b[2]; uint32_t u; } pk;
        pk.b[0] = (__bf16)(qkv_w[n * 512 + k] * s);
        pk.b[1] = (__bf16)(qkv_w[n * 512 + k + 1] * s);
        wq[idx] = pk.u;
    } else if (idx < WQ_DWORDS + WP_DWORDS) {
        int i2 = idx - WQ_DWORDS;
        int v = i2 & 7, lane = (i2 >> 3) & 31, t = i2 >> 8;
        int tj = t & 31, kt = t >> 5;
        int n = tj * 16 + (lane & 15);
        int k = kt * 32 + ((lane & 16) ? 8 : 0) + ((v & 4) ? 16 : 0) + ((v & 3) << 1);
        union { __bf16 b[2]; uint32_t u; } pk;
        pk.b[0] = (__bf16)proj_w[n * 512 + k];
        pk.b[1] = (__bf16)proj_w[n * 512 + k + 1];
        wp[i2] = pk.u;
    }
}

// ---------------------------------------------------------------------------
// Main kernel: 1 block per window, 256 threads = 8 waves.
// Dynamic LDS layout (327680 B = 320 KB total):
//   [0      , 65536 ) sX   : x window, bf16 [64][512]; reused as per-wave P bufs
//   [65536  , 131072) sQ   : bf16 [16 heads][64 rows][32]
//   [131072 , 196608) sK   : bf16 [16][64][32]
//   [196608 , 262144) sVt  : bf16 [16][32 d][64 keys]  (transposed V)
//   [262144 , 327680) sO   : bf16 [64][512]  attention output (proj input)
// ---------------------------------------------------------------------------
__global__ __launch_bounds__(256) void swin_attn_kernel(
        const float* __restrict__ x,          // [8,4096,512]
        const float* __restrict__ qkv_bias,   // [1536]
        const float* __restrict__ proj_bias,  // [512]
        const float* __restrict__ rpb,        // [127,16]
        const uint32_t* __restrict__ wq,      // packed qkv weights
        const uint32_t* __restrict__ wp,      // packed proj weights
        float* __restrict__ out) {            // [8,4096,512]
    extern __shared__ char smem[];
    __bf16* sX  = (__bf16*)(smem);
    __bf16* sQ  = (__bf16*)(smem + 65536);
    __bf16* sK  = (__bf16*)(smem + 131072);
    __bf16* sVt = (__bf16*)(smem + 196608);
    __bf16* sO  = (__bf16*)(smem + 262144);

    const int tid  = threadIdx.x;
    const int wave = tid >> 5;
    const int lane = tid & 31;
    const int ln16 = lane & 15;
    const int hb   = (lane & 16) ? 8 : 0;   // row-half offset in C-layout / K-half in A-layout

    const int blk = blockIdx.x;
    const int bat = blk >> 6;               // batch index
    const int win = blk & 63;               // window index within batch
    const bool last_win = (win == 63);      // only window with non-zero shift mask

    // ---- phase 0: load rolled window into LDS as bf16 (float4 granules) --
    {
        const int row0 = win * 64;
        // 64*512 floats = 8192 float4; 32 per thread
        for (int i = tid; i < 8192; i += 256) {
            int m = i >> 7;                 // row (128 float4 per row)
            int c4 = i & 127;               // float4 index within row
            int src = (row0 + m + 32) & 4095;            // xp = roll(x, -32)
            f32x4 f = *(const f32x4*)(x + ((size_t)bat * 4096 + src) * 512 + c4 * 4);
            union { __bf16 b[4]; uint2 u; } pk;
            pk.b[0] = (__bf16)f.x; pk.b[1] = (__bf16)f.y;
            pk.b[2] = (__bf16)f.z; pk.b[3] = (__bf16)f.w;
            *(uint2*)(sX + m * 512 + c4 * 4) = pk.u;
        }
    }
    __syncthreads();

    // ---- phase 1: QKV = sX @ Wqkv^T  (64x512 @ 512x1536) -----------------
    {
        const int ti = wave & 3;            // row tile 0..3
        const int ch = wave >> 2;           // column half 0..1 (48 col-tiles each)
        const __bf16* aBase = sX + (ti * 16 + ln16) * 512 + hb;       // + kt*32
        for (int chunk = 0; chunk < 6; ++chunk) {
            const int tj0 = ch * 48 + chunk * 8;
            // single 64-bit base per chunk; all fragment addressing via
            // compile-time byte offsets folded into the load immediates
            const char* bb = (const char*)wq + ((size_t)(tj0 * 32 + lane)) * 32;
            f32x8 acc[8] = {};
#pragma unroll 2
            for (int kt = 0; kt < 16; ++kt) {
                Frag a;
                a.h.lo = *(const bf16x8*)(aBase + kt * 32);
                a.h.hi = *(const bf16x8*)(aBase + kt * 32 + 16);
                const char* bk = bb + kt * (96 * 1024);   // k-step stride 96 KB
#pragma unroll
                for (int j = 0; j < 8; ++j) {
                    Frag bw;
                    bw.h.lo = *(const bf16x8*)(bk + j * 1024);
                    bw.h.hi = *(const bf16x8*)(bk + j * 1024 + 16);
                    acc[j] = wmma_bf16(a, bw, acc[j]);
                }
            }
            // bias + bf16 convert + scatter into head-major q/k/v layouts
#pragma unroll
            for (int j = 0; j < 8; ++j) {
                const int n = (tj0 + j) * 16 + ln16;      // output column 0..1535
                const int s = n >> 9, h = (n >> 5) & 15, d = n & 31;
                const float bias = qkv_bias[n] * (s == 0 ? QSCALE : 1.0f);
#pragma unroll
                for (int r = 0; r < 8; ++r) {
                    const int m = ti * 16 + r + hb;
                    const float val = acc[j][r] + bias;
                    if (s == 0)      sQ[(h * 64 + m) * 32 + d]  = (__bf16)val;
                    else if (s == 1) sK[(h * 64 + m) * 32 + d]  = (__bf16)val;
                    else             sVt[(h * 32 + d) * 64 + m] = (__bf16)val; // transposed
                }
            }
        }
    }
    __syncthreads();

    // ---- phase 2: attention, 2 heads per wave ----------------------------
    {
        __bf16* P = sX + wave * 4096;      // per-wave [64][64] bf16 prob buffer
        for (int it = 0; it < 2; ++it) {
            const int h = wave * 2 + it;
            // K fragments (B operand of Q@K^T), reused across row tiles
            Frag kb[4];
            const __bf16* kBase = sK + (h * 64 + ln16) * 32 + hb;     // + tj*16*32
#pragma unroll
            for (int tj = 0; tj < 4; ++tj) {
                kb[tj].h.lo = *(const bf16x8*)(kBase + tj * 512);
                kb[tj].h.hi = *(const bf16x8*)(kBase + tj * 512 + 16);
            }
            // S = Q K^T, bias + mask, row softmax, write P (bf16)
            const __bf16* qBase = sQ + (h * 64 + ln16) * 32 + hb;     // + ti*16*32
            for (int ti = 0; ti < 4; ++ti) {
                Frag qa;
                qa.h.lo = *(const bf16x8*)(qBase + ti * 512);
                qa.h.hi = *(const bf16x8*)(qBase + ti * 512 + 16);
                f32x8 st[4] = {};
#pragma unroll
                for (int tj = 0; tj < 4; ++tj)
                    st[tj] = wmma_bf16(qa, kb[tj], st[tj]);
#pragma unroll
                for (int r = 0; r < 8; ++r) {
                    const int q = ti * 16 + r + hb;
                    float vals[4];
#pragma unroll
                    for (int tj = 0; tj < 4; ++tj) {
                        const int kc = tj * 16 + ln16;
                        float v = st[tj][r] + rpb[(q - kc + 63) * 16 + h];
                        if (last_win && (((q ^ kc) & 32) != 0)) v += -100.0f;
                        vals[tj] = v;
                    }
                    float mx = fmaxf(fmaxf(vals[0], vals[1]), fmaxf(vals[2], vals[3]));
                    mx = fmaxf(mx, __shfl_xor(mx, 1));
                    mx = fmaxf(mx, __shfl_xor(mx, 2));
                    mx = fmaxf(mx, __shfl_xor(mx, 4));
                    mx = fmaxf(mx, __shfl_xor(mx, 8));
                    float e[4], sum = 0.0f;
#pragma unroll
                    for (int tj = 0; tj < 4; ++tj) { e[tj] = __expf(vals[tj] - mx); sum += e[tj]; }
                    sum += __shfl_xor(sum, 1);
                    sum += __shfl_xor(sum, 2);
                    sum += __shfl_xor(sum, 4);
                    sum += __shfl_xor(sum, 8);
                    const float inv = __builtin_amdgcn_rcpf(sum);
#pragma unroll
                    for (int tj = 0; tj < 4; ++tj)
                        P[q * 64 + tj * 16 + ln16] = (__bf16)(e[tj] * inv);
                }
            }
            // O = P @ V  (64x64 @ 64x32), V read transposed -> contiguous K
            Frag vb[4];   // [tj(2)][kt(2)]
            const __bf16* vBase = sVt + (h * 32 + ln16) * 64 + hb;    // + tj*16*64 + kt*32
#pragma unroll
            for (int tj = 0; tj < 2; ++tj)
#pragma unroll
                for (int kt = 0; kt < 2; ++kt) {
                    const __bf16* vp = vBase + tj * 1024 + kt * 32;
                    vb[tj * 2 + kt].h.lo = *(const bf16x8*)vp;
                    vb[tj * 2 + kt].h.hi = *(const bf16x8*)(vp + 16);
                }
            const __bf16* pBase = P + ln16 * 64 + hb;                 // + ti*16*64 + kt*32
            for (int ti = 0; ti < 4; ++ti) {
                Frag pa[2];
#pragma unroll
                for (int kt = 0; kt < 2; ++kt) {
                    const __bf16* pp = pBase + ti * 1024 + kt * 32;
                    pa[kt].h.lo = *(const bf16x8*)pp;
                    pa[kt].h.hi = *(const bf16x8*)(pp + 16);
                }
#pragma unroll
                for (int tj = 0; tj < 2; ++tj) {
                    f32x8 o = {};
                    o = wmma_bf16(pa[0], vb[tj * 2 + 0], o);
                    o = wmma_bf16(pa[1], vb[tj * 2 + 1], o);
#pragma unroll
                    for (int r = 0; r < 8; ++r) {
                        const int m = ti * 16 + r + hb;
                        sO[m * 512 + h * 32 + tj * 16 + ln16] = (__bf16)o[r];
                    }
                }
            }
        }
    }
    __syncthreads();

    // ---- phase 3: proj GEMM (64x512 @ 512x512) + bias + inverse roll -----
    {
        const int ti = wave & 3;
        const int ch = wave >> 2;
        const __bf16* aBase = sO + (ti * 16 + ln16) * 512 + hb;       // + kt*32
        for (int chunk = 0; chunk < 2; ++chunk) {
            const int tj0 = ch * 16 + chunk * 8;
            const char* bb = (const char*)wp + ((size_t)(tj0 * 32 + lane)) * 32;
            f32x8 acc[8] = {};
#pragma unroll 2
            for (int kt = 0; kt < 16; ++kt) {
                Frag a;
                a.h.lo = *(const bf16x8*)(aBase + kt * 32);
                a.h.hi = *(const bf16x8*)(aBase + kt * 32 + 16);
                const char* bk = bb + kt * (32 * 1024);   // k-step stride 32 KB
#pragma unroll
                for (int j = 0; j < 8; ++j) {
                    Frag bw;
                    bw.h.lo = *(const bf16x8*)(bk + j * 1024);
                    bw.h.hi = *(const bf16x8*)(bk + j * 1024 + 16);
                    acc[j] = wmma_bf16(a, bw, acc[j]);
                }
            }
#pragma unroll
            for (int j = 0; j < 8; ++j) {
                const int n = (tj0 + j) * 16 + ln16;
                const float bias = proj_bias[n];
#pragma unroll
                for (int r = 0; r < 8; ++r) {
                    const int m = ti * 16 + r + hb;
                    const int dst = (win * 64 + m + 32) & 4095;   // roll(out, +32)
                    out[((size_t)bat * 4096 + dst) * 512 + n] = acc[j][r] + bias;
                }
            }
        }
    }
}

// ---------------------------------------------------------------------------
extern "C" void kernel_launch(void* const* d_in, const int* in_sizes, int n_in,
                              void* d_out, int out_size, void* d_ws, size_t ws_size,
                              hipStream_t stream) {
    const float* x   = (const float*)d_in[0];   // [8,4096,512]
    const float* qw  = (const float*)d_in[1];   // [1536,512]
    const float* qb  = (const float*)d_in[2];   // [1536]
    const float* pw  = (const float*)d_in[3];   // [512,512]
    const float* pb  = (const float*)d_in[4];   // [512]
    const float* rpb = (const float*)d_in[5];   // [127,16]

    uint32_t* wq = (uint32_t*)d_ws;             // 1.5 MB packed qkv weights
    uint32_t* wp = wq + WQ_DWORDS;              // 0.5 MB packed proj weights

    // allow 320 KB dynamic LDS on the attention kernel
    static const size_t kLds = 327680;
    hipFuncSetAttribute((const void*)swin_attn_kernel,
                        hipFuncAttributeMaxDynamicSharedMemorySize, (int)kLds);

    const int packN = WQ_DWORDS + WP_DWORDS;
    pack_weights_kernel<<<(packN + 255) / 256, 256, 0, stream>>>(qw, pw, wq, wp);
    swin_attn_kernel<<<512, 256, kLds, stream>>>(x, qb, pb, rpb, wq, wp, (float*)d_out);
}
